// ExpertLayer_48619029791273
// MI455X (gfx1250) — matrix operand order repaired and checked by reading, not compiled, not run on hardware
//
#include <hip/hip_runtime.h>
#include <hip/hip_bf16.h>
#include <math.h>

// ---------------- problem constants (match reference) ----------------
#define NE   8
#define TOPK 2
#define HD   1024
#define FD   4096
#define NB   2
#define NS   2048
#define NTOK (NB*NS)          // 4096
#define CAP  2048             // int(N/E * 4.0)
#define LB_COEF 0.001f

typedef unsigned short u16;
typedef unsigned int   u32;
typedef __attribute__((ext_vector_type(16))) __bf16 v16bf;
typedef __attribute__((ext_vector_type(8)))  float  v8f;
typedef __attribute__((ext_vector_type(8)))  u32    v8u;
typedef __attribute__((ext_vector_type(4)))  u32    v4u;

static __device__ __forceinline__ u16 f32_to_bf16(float f) {
    u32 u = __float_as_uint(f);
    u = u + 0x7FFFu + ((u >> 16) & 1u);   // round-to-nearest-even
    return (u16)(u >> 16);
}
static __device__ __forceinline__ u32 pack_bf16x2(float lo, float hi) {
    return (u32)f32_to_bf16(lo) | ((u32)f32_to_bf16(hi) << 16);
}
static __device__ __forceinline__ v8f zero8() {
    v8f z;
#pragma unroll
    for (int i = 0; i < 8; ++i) z[i] = 0.0f;
    return z;
}
static __device__ __forceinline__ v16bf frag_from(v4u a, v4u b) {
    v8u u;
    u[0]=a[0]; u[1]=a[1]; u[2]=a[2]; u[3]=a[3];
    u[4]=b[0]; u[5]=b[1]; u[6]=b[2]; u[7]=b[3];
    return __builtin_bit_cast(v16bf, u);
}
// generic LDS pointer -> 32-bit DS byte offset (low 32 bits of aperture addr)
static __device__ __forceinline__ u32 lds_off(const void* p) {
    return (u32)(size_t)p;
}

// ---------------- tiny init ----------------
__global__ void moe_zero_kernel(int* cnt, float* f_cnt, float* p_sum) {
    int t = threadIdx.x;
    if (t < NE) { cnt[t] = 0; f_cnt[t] = 0.0f; p_sum[t] = 0.0f; }
}

// ---------------- router: one block per token ----------------
__global__ __launch_bounds__(256)
void moe_router_kernel(const float* __restrict__ x, const float* __restrict__ Wr,
                       const float* __restrict__ br,
                       int* __restrict__ cnt, float* __restrict__ f_cnt,
                       float* __restrict__ p_sum,
                       int* __restrict__ disp, int* __restrict__ tok_e,
                       int* __restrict__ tok_s, float* __restrict__ tok_w) {
    const int n    = blockIdx.x;
    const int lane = threadIdx.x & 31;
    const int wave = threadIdx.x >> 5;        // 8 waves == 8 experts
    const float* xr = x  + (size_t)n    * HD;
    const float* wr = Wr + (size_t)wave * HD;
    float s = 0.0f;
    for (int h = lane; h < HD; h += 32) s += xr[h] * wr[h];
#pragma unroll
    for (int off = 16; off > 0; off >>= 1) s += __shfl_xor(s, off, 32);
    __shared__ float lg[NE];
    if (lane == 0) lg[wave] = s + br[wave];
    __syncthreads();
    if (threadIdx.x == 0) {
        float l[NE];
#pragma unroll
        for (int e = 0; e < NE; ++e) l[e] = lg[e];
        int i0 = 0;
        for (int e = 1; e < NE; ++e) if (l[e] > l[i0]) i0 = e;
        int i1 = -1;
        for (int e = 0; e < NE; ++e) {
            if (e == i0) continue;
            if (i1 < 0 || l[e] > l[i1]) i1 = e;
        }
        float e0 = 1.0f;                       // expf(l[i0]-l[i0])
        float e1 = expf(l[i1] - l[i0]);
        float inv = 1.0f / (e0 + e1);
        float w0 = e0 * inv, w1 = e1 * inv;
        float mx = l[0];
        for (int e = 1; e < NE; ++e) mx = fmaxf(mx, l[e]);
        float pe[NE], se = 0.0f;
#pragma unroll
        for (int e = 0; e < NE; ++e) { pe[e] = expf(l[e] - mx); se += pe[e]; }
        float ise = 1.0f / se;
#pragma unroll
        for (int e = 0; e < NE; ++e) atomicAdd(&p_sum[e], pe[e] * ise);
        atomicAdd(&f_cnt[i0], 1.0f);
        int   ee[2] = { i0, i1 };
        float ww[2] = { w0, w1 };
#pragma unroll
        for (int k = 0; k < TOPK; ++k) {
            int e = ee[k];
            int pos = atomicAdd(&cnt[e], 1);
            int slot = (pos < CAP) ? pos : -1;
            if (slot >= 0) disp[e * CAP + slot] = n;
            tok_e[n * TOPK + k] = e;
            tok_s[n * TOPK + k] = slot;
            tok_w[n * TOPK + k] = ww[k];
        }
    }
}

__global__ void moe_aux_kernel(const float* f_cnt, const float* p_sum, float* aux_out) {
    if (threadIdx.x == 0) {
        const float invN = 1.0f / (float)NTOK;
        float s = 0.0f;
        for (int e = 0; e < NE; ++e) s += (f_cnt[e] * invN) * (p_sum[e] * invN);
        *aux_out = (float)NE * s * LB_COEF;
    }
}

// ---------------- fp32 -> bf16 elementwise (weights), float4 vectorized ----------------
__global__ void moe_cvt_bf16_kernel(const float* __restrict__ src,
                                    u32* __restrict__ dst, size_t n4) {
    size_t i = (size_t)blockIdx.x * blockDim.x + threadIdx.x;
    size_t stride = (size_t)gridDim.x * blockDim.x;
    for (; i < n4; i += stride) {
        float4 v = reinterpret_cast<const float4*>(src)[i];
        uint2 o;
        o.x = pack_bf16x2(v.x, v.y);
        o.y = pack_bf16x2(v.z, v.w);
        reinterpret_cast<uint2*>(dst)[i] = o;
    }
}

// ---------------- gather dispatched token rows -> bf16 ----------------
__global__ __launch_bounds__(256)
void moe_gather_kernel(const float* __restrict__ x, const int* __restrict__ disp,
                       const int* __restrict__ cnt, u32* __restrict__ expert_in) {
    int e = blockIdx.y, slot = blockIdx.x;
    if (slot >= cnt[e]) return;
    int tok = disp[e * CAP + slot];
    const float4* src = reinterpret_cast<const float4*>(x + (size_t)tok * HD);
    uint2* dst = reinterpret_cast<uint2*>(expert_in + ((size_t)e * CAP + slot) * (HD / 2));
    for (int h = threadIdx.x; h < HD / 4; h += 256) {
        float4 v = src[h];
        uint2 o;
        o.x = pack_bf16x2(v.x, v.y);
        o.y = pack_bf16x2(v.z, v.w);
        dst[h] = o;
    }
}

// ---------------- tiled NT GEMM on WMMA bf16 + async-to-LDS + b128 LDS reads ----------------
// A: [E][M][K] bf16 row-major, Bw: [E][N][K] bf16 row-major (weights, K inner)
// EPI==0: out = bf16( gelu_exact(acc + bias) )   (h buffer)
// EPI==1: out = f32 ( acc + bias )               (expert_out)
// Pipeline is unrolled x2 so the double-buffer index is a compile-time constant.
#define TSTR 40   // LDS row stride in u16 (80B = 5*16B: aligned, bank-spread)

template <int EPI>
__global__ __launch_bounds__(256)
void moe_gemm_kernel(const u16* __restrict__ A, const u16* __restrict__ Bw,
                     const float* __restrict__ bias,
                     u16* __restrict__ Obf, float* __restrict__ Of,
                     const int* __restrict__ cnt, int M, int N, int K) {
    const int e  = blockIdx.z;
    const int mB = blockIdx.x * 128;
    const int nB = blockIdx.y * 128;
    if (mB >= cnt[e]) return;                 // whole tile beyond routed count

    __shared__ u16 As[2][128 * TSTR];         // token tile,   [m][k] row-major
    __shared__ u16 Bs[2][128 * TSTR];         // weight tile,  [n][k] row-major

    const u16* Ae = A  + ((size_t)e * M + mB) * K;
    const u16* Be = Bw + ((size_t)e * N + nB) * K;

    const int tid  = threadIdx.x;
    const int lane = tid & 31;
    const int wave = tid >> 5;
    const int wm   = wave & 3;                // 4 waves over M (32 rows each)
    const int wn   = wave >> 2;               // 2 waves over N (64 cols each)

    const int srow  = tid >> 1;               // 0..127 staging row
    const int shalf = (tid & 1) * 16;         // 0/16 u16 = 0/32B k-offset

    u32 aDst[2], bDst[2];
#pragma unroll
    for (int b = 0; b < 2; ++b) {
        aDst[b] = lds_off(&As[b][srow * TSTR + shalf]);
        bDst[b] = lds_off(&Bs[b][srow * TSTR + shalf]);
    }
    // per-lane fragment base addresses (bytes); A and B share the same form:
    // row = (lane&15) within 16-tile, k-run byte offset = 16*(lane>=16)
    const u32 aFrag0 = lds_off(&As[0][0]) + (u32)(((wm * 32 + (lane & 15)) * TSTR) * 2) + ((lane >> 4) * 16);
    const u32 bFrag0 = lds_off(&Bs[0][0]) + (u32)(((wn * 64 + (lane & 15)) * TSTR) * 2) + ((lane >> 4) * 16);
    const u32 bufStep = (u32)(128 * TSTR * 2);

    const u16* Arow = Ae + (size_t)srow * K + shalf;
    const u16* Brow = Be + (size_t)srow * K + shalf;

    v8f acc[2][4];
#pragma unroll
    for (int mi = 0; mi < 2; ++mi)
#pragma unroll
        for (int ni = 0; ni < 4; ++ni) acc[mi][ni] = zero8();

    auto stage = [&](u32 aD, u32 bD, int kb) {
        unsigned long long ga = (unsigned long long)(const void*)(Arow + kb);
        unsigned long long gb = (unsigned long long)(const void*)(Brow + kb);
        asm volatile("global_load_async_to_lds_b128 %0, %1, off\n\t"
                     "global_load_async_to_lds_b128 %2, %3, off"
                     :: "v"(aD), "v"(ga), "v"(bD), "v"(gb) : "memory");
    };
    auto wait_barrier = [&]() {
        asm volatile("s_wait_asynccnt 0x0" ::: "memory");
        __syncthreads();
    };
    auto compute = [&](u32 aF, u32 bF) {
        // B fragments: plain b128 reads of the two 8-element k-runs (+0B / +32B)
        v4u btr[4][2];
#pragma unroll
        for (int ni = 0; ni < 4; ++ni) {
            u32 baddr = bF + (u32)(ni * 16 * TSTR * 2);
            asm volatile("ds_load_b128 %0, %2\n\t"
                         "ds_load_b128 %1, %2 offset:32"
                         : "=v"(btr[ni][0]), "=v"(btr[ni][1]) : "v"(baddr) : "memory");
        }
        // A fragments: plain b128 reads of the two 8-element k-runs
        v4u atr[2][2];
#pragma unroll
        for (int mi = 0; mi < 2; ++mi) {
            u32 aaddr = aF + (u32)(mi * 16 * TSTR * 2);
            asm volatile("ds_load_b128 %0, %2\n\t"
                         "ds_load_b128 %1, %2 offset:32"
                         : "=v"(atr[mi][0]), "=v"(atr[mi][1]) : "v"(aaddr) : "memory");
        }
        asm volatile("s_wait_dscnt 0x0" ::: "memory");

        v16bf af[2], bf[4];
#pragma unroll
        for (int mi = 0; mi < 2; ++mi) af[mi] = frag_from(atr[mi][0], atr[mi][1]);
#pragma unroll
        for (int ni = 0; ni < 4; ++ni) bf[ni] = frag_from(btr[ni][0], btr[ni][1]);

#pragma unroll
        for (int mi = 0; mi < 2; ++mi)
#pragma unroll
            for (int ni = 0; ni < 4; ++ni)
                acc[mi][ni] = __builtin_amdgcn_wmma_f32_16x16x32_bf16(
                    false, af[mi], false, bf[ni], (short)0, acc[mi][ni], false, false);
    };

    // ---- software pipeline, x2-unrolled: buffer index is static ----
    stage(aDst[0], bDst[0], 0);
    for (int kb = 0; kb < K; kb += 64) {
        wait_barrier();                               // buf0 of this pair is ready
        if (kb + 32 < K) stage(aDst[1], bDst[1], kb + 32);
        compute(aFrag0, bFrag0);                      // consume buf0
        wait_barrier();                               // buf1 ready
        if (kb + 64 < K) stage(aDst[0], bDst[0], kb + 64);
        compute(aFrag0 + bufStep, bFrag0 + bufStep);  // consume buf1
    }

    // epilogue: C/D layout — lanes 0-15: n=lane, m=v; lanes 16-31: n=lane-16, m=8+v
    const int nloc = lane & 15;
    const int mh   = (lane >> 4) * 8;
#pragma unroll
    for (int mi = 0; mi < 2; ++mi) {
#pragma unroll
        for (int ni = 0; ni < 4; ++ni) {
            const int gm0 = mB + wm * 32 + mi * 16 + mh;
            const int gn  = nB + wn * 64 + ni * 16 + nloc;
            const float bb = bias[e * N + gn];
#pragma unroll
            for (int v = 0; v < 8; ++v) {
                const int gm = gm0 + v;
                float val = acc[mi][ni][v] + bb;
                if (EPI == 0) {
                    float g = 0.5f * val * (1.0f + erff(val * 0.70710678118654752f));
                    Obf[((size_t)e * M + gm) * N + gn] = f32_to_bf16(g);
                } else {
                    Of[((size_t)e * M + gm) * N + gn] = val;
                }
            }
        }
    }
}

// ---------------- combine: 2 fixed-order terms per token ----------------
__global__ __launch_bounds__(256)
void moe_combine_kernel(const float* __restrict__ eout, const int* __restrict__ tok_e,
                        const int* __restrict__ tok_s, const float* __restrict__ tok_w,
                        float* __restrict__ out) {
    int n = blockIdx.x;
    int e0 = tok_e[n * 2], e1 = tok_e[n * 2 + 1];
    int s0 = tok_s[n * 2], s1 = tok_s[n * 2 + 1];
    float w0 = (s0 >= 0) ? tok_w[n * 2]     : 0.0f;
    float w1 = (s1 >= 0) ? tok_w[n * 2 + 1] : 0.0f;
    const float4* r0 = reinterpret_cast<const float4*>(
        eout + ((size_t)e0 * CAP + (s0 < 0 ? 0 : s0)) * HD);
    const float4* r1 = reinterpret_cast<const float4*>(
        eout + ((size_t)e1 * CAP + (s1 < 0 ? 0 : s1)) * HD);
    float4* o = reinterpret_cast<float4*>(out + (size_t)n * HD);
    for (int h = threadIdx.x; h < HD / 4; h += 256) {
        float4 a = r0[h], b = r1[h], v;
        v.x = w0 * a.x + w1 * b.x;
        v.y = w0 * a.y + w1 * b.y;
        v.z = w0 * a.z + w1 * b.z;
        v.w = w0 * a.w + w1 * b.w;
        o[h] = v;
    }
}

// ---------------- host side ----------------
static inline size_t align256(size_t x) { return (x + 255) & ~(size_t)255; }

extern "C" void kernel_launch(void* const* d_in, const int* in_sizes, int n_in,
                              void* d_out, int out_size, void* d_ws, size_t ws_size,
                              hipStream_t stream) {
    const float* x   = (const float*)d_in[0];
    const float* Wr  = (const float*)d_in[1];
    const float* br  = (const float*)d_in[2];
    const float* Wup = (const float*)d_in[3];
    const float* bup = (const float*)d_in[4];
    const float* Wdn = (const float*)d_in[5];
    const float* bdn = (const float*)d_in[6];
    float* out = (float*)d_out;                 // [N*H] + [1] aux

    char* p = (char*)d_ws;
    size_t off = 0;
    auto take = [&](size_t bytes) { void* r = p + off; off = align256(off + bytes); return r; };
    int*   cnt   = (int*)  take(NE * sizeof(int));
    float* f_cnt = (float*)take(NE * sizeof(float));
    float* p_sum = (float*)take(NE * sizeof(float));
    int*   disp  = (int*)  take((size_t)NE * CAP * sizeof(int));
    int*   tok_e = (int*)  take((size_t)NTOK * TOPK * sizeof(int));
    int*   tok_s = (int*)  take((size_t)NTOK * TOPK * sizeof(int));
    float* tok_w = (float*)take((size_t)NTOK * TOPK * sizeof(float));
    u16*   xin16 = (u16*)  take((size_t)NE * CAP * HD * sizeof(u16));
    u16*   wup16 = (u16*)  take((size_t)NE * FD * HD * sizeof(u16));
    u16*   wdn16 = (u16*)  take((size_t)NE * HD * FD * sizeof(u16));
    u16*   h16   = (u16*)  take((size_t)NE * CAP * FD * sizeof(u16));
    float* eout  = (float*)take((size_t)NE * CAP * HD * sizeof(float));
    (void)ws_size; (void)n_in; (void)in_sizes; (void)out_size;

    moe_zero_kernel<<<1, 32, 0, stream>>>(cnt, f_cnt, p_sum);

    moe_router_kernel<<<NTOK, 256, 0, stream>>>(x, Wr, br, cnt, f_cnt, p_sum,
                                                disp, tok_e, tok_s, tok_w);

    moe_aux_kernel<<<1, 32, 0, stream>>>(f_cnt, p_sum, out + (size_t)NTOK * HD);

    const size_t wn4 = (size_t)NE * FD * HD / 4;
    moe_cvt_bf16_kernel<<<4096, 256, 0, stream>>>(Wup, (u32*)wup16, wn4);
    moe_cvt_bf16_kernel<<<4096, 256, 0, stream>>>(Wdn, (u32*)wdn16, wn4);

    moe_gather_kernel<<<dim3(CAP, NE), 256, 0, stream>>>(x, disp, cnt, (u32*)xin16);

    // GEMM1: [cap x H] x [F x H]^T -> gelu -> h (bf16)
    moe_gemm_kernel<0><<<dim3(CAP / 128, FD / 128, NE), 256, 0, stream>>>(
        xin16, wup16, bup, h16, nullptr, cnt, CAP, FD, HD);

    // GEMM2: [cap x F] x [H x F]^T -> expert_out (f32)
    moe_gemm_kernel<1><<<dim3(CAP / 128, HD / 128, NE), 256, 0, stream>>>(
        h16, wdn16, bdn, nullptr, eout, cnt, CAP, HD, FD);

    moe_combine_kernel<<<NTOK, 256, 0, stream>>>(eout, tok_e, tok_s, tok_w, out);
}
